// Gemma4TextAttention_78030965834334
// MI455X (gfx1250) — matrix operand order repaired and checked by reading, not compiled, not run on hardware
//
#include <hip/hip_runtime.h>
#include <hip/hip_bf16.h>

// ---------------------------------------------------------------------------
// Gemma attention for MI455X (gfx1250, wave32, WMMA).
// All matmuls run on V_WMMA_F32_16X16X32_BF16: fp32 data is loaded from HBM
// (bandwidth floor ~128MB @ 23.3TB/s ~ 5.5us), converted to bf16 with packed
// v_cvt_pk_bf16_f32, accumulated in f32. Each wave produces a 16x32 output
// tile (2 WMMAs sharing one A operand). GEMM shapes are template constants so
// all strided-row addresses fold into 24-bit instruction offsets (no spills).
// KV-cache roll handled by index math (no 64MB copy).
// ---------------------------------------------------------------------------

#define H_    8
#define KV_   4
#define G_    2
#define D_    256
#define HID_  2560
#define S_    32
#define L_    8192
#define EPS_  1e-6f

typedef __attribute__((ext_vector_type(16))) __bf16 v16bf;
typedef __attribute__((ext_vector_type(8)))  float  v8f;

// Native f32 -> bf16 (RTNE) — lowers to packed v_cvt_pk_bf16_f32.
__device__ __forceinline__ v16bf cvt16(const float* f) {
  v16bf v;
#pragma unroll
  for (int j = 0; j < 16; ++j) v[j] = (__bf16)f[j];
  return v;
}

__device__ __forceinline__ v8f wmma_bf16(v16bf a, v16bf b, v8f c) {
  return __builtin_amdgcn_wmma_f32_16x16x32_bf16(false, a, false, b,
                                                 (short)0, c, false, false);
}

// ---------------------------------------------------------------------------
// Generic GEMM: C[M,N] = A[M,K] * B[K,N], f32 in memory, bf16 WMMA inside.
// N, K compile-time so B row offsets (j*N*4 <= 153KB) fold into the 24-bit
// GLOBAL_LOAD immediate — one live base pointer per operand.
// One wave per 16x32 C tile (2 accumulators, shared A). grid=(N/32, M/16).
// A layout (16-bit 16x32): lane m = lane&15; lanes<16 take K 0-15 of
// each 32-chunk, lanes>=16 take K 16-31.
// B layout: lane n = lane&15; lanes<16 hold K 0-15, lanes>=16 hold K 16-31.
// ---------------------------------------------------------------------------
template <int N, int K>
__global__ __launch_bounds__(32) void gemm_bf16wmma(const float* __restrict__ A,
                                                    const float* __restrict__ B,
                                                    float* __restrict__ C) {
  const int lane  = threadIdx.x & 31;
  const int ntile = blockIdx.x, mtile = blockIdx.y;
  const int col0  = ntile * 32 + (lane & 15);
  const int base8 = (lane < 16) ? 0 : 16;
  const int koff  = (lane < 16) ? 0 : 16;
  const int m     = mtile * 16 + (lane & 15);

  const float* aptr = A + (size_t)m * K + base8;
  const float* bptr = B + (size_t)koff * N + col0;

  v8f acc0 = {0.f, 0.f, 0.f, 0.f, 0.f, 0.f, 0.f, 0.f};
  v8f acc1 = acc0;
#pragma unroll 2
  for (int kb = 0; kb < K; kb += 32) {
    float af[16], bf0[16], bf1[16];
    *(float4*)(af + 0)  = *(const float4*)(aptr + 0);
    *(float4*)(af + 4)  = *(const float4*)(aptr + 4);
    *(float4*)(af + 8)  = *(const float4*)(aptr + 8);
    *(float4*)(af + 12) = *(const float4*)(aptr + 12);
#pragma unroll
    for (int j = 0; j < 16; ++j) {           // j*N folds into imm offset
      bf0[j] = bptr[j * N];
      bf1[j] = bptr[j * N + 16];
    }
    __builtin_prefetch(aptr + 160, 0, 1);    // global_prefetch_b8 (spec, OOB ok)
    v16bf a = cvt16(af);
    acc0 = wmma_bf16(a, cvt16(bf0), acc0);
    acc1 = wmma_bf16(a, cvt16(bf1), acc1);
    aptr += 32;
    bptr += 32 * N;
  }
  // C/D layout: VGPR r -> row r (lanes 0-15) / row 8+r (lanes 16-31)
  const int rbase = mtile * 16 + ((lane >> 4) << 3);
  float* cptr = C + (size_t)rbase * N + col0;
#pragma unroll
  for (int r = 0; r < 8; ++r) {              // r*N folds into imm offset
    cptr[r * N]      = acc0[r];
    cptr[r * N + 16] = acc1[r];
  }
}

// ---------------------------------------------------------------------------
// RMSNorm (+RoPE for q/k): one wave per row of D=256 (8 f32 per lane).
// RoPE partner element d^128 lives at lane^16 -> single wave32 shuffle.
// Rows 0..255 = q (regrouped to [KV][G*S][D]); 256..383 = k_new; 384..511 = v_new.
// ---------------------------------------------------------------------------
__global__ __launch_bounds__(256) void norm_rope_kernel(
    const float* __restrict__ q_lin, const float* __restrict__ k_lin,
    const float* __restrict__ v_lin, const float* __restrict__ cos_t,
    const float* __restrict__ sin_t, const float* __restrict__ qw,
    const float* __restrict__ kw, const float* __restrict__ vw,
    float* __restrict__ q_g, float* __restrict__ k_new,
    float* __restrict__ v_new) {
  const int lane = threadIdx.x & 31;
  const int wid  = threadIdx.x >> 5;
  const int row  = blockIdx.x * 8 + wid;          // 0..511
  const int d0   = lane * 8;

  const float* src; const float* w; float* dst; int s; bool do_rope;
  if (row < 256) {                                 // Q
    int h = row >> 5; s = row & 31;
    src = q_lin + (size_t)s * (H_ * D_) + h * D_;
    w = qw; do_rope = true;
    int kv = h >> 1, g = h & 1;
    dst = q_g + (size_t)(kv * (G_ * S_) + g * S_ + s) * D_;
  } else if (row < 384) {                          // K
    int i = row - 256; int kv = i >> 5; s = i & 31;
    src = k_lin + (size_t)s * (KV_ * D_) + kv * D_;
    w = kw; do_rope = true;
    dst = k_new + (size_t)(kv * S_ + s) * D_;
  } else {                                         // V
    int i = row - 384; int kv = i >> 5; s = i & 31;
    src = v_lin + (size_t)s * (KV_ * D_) + kv * D_;
    w = vw; do_rope = false;
    dst = v_new + (size_t)(kv * S_ + s) * D_;
  }

  float x[8];
  *(float4*)(x)     = *(const float4*)(src + d0);
  *(float4*)(x + 4) = *(const float4*)(src + d0 + 4);
  float ss = 0.f;
#pragma unroll
  for (int j = 0; j < 8; ++j) ss += x[j] * x[j];
#pragma unroll
  for (int msk = 16; msk; msk >>= 1) ss += __shfl_xor(ss, msk, 32);
  const float scale = rsqrtf(ss * (1.0f / D_) + EPS_);

  float xn[8];
#pragma unroll
  for (int j = 0; j < 8; ++j) xn[j] = x[j] * scale * w[d0 + j];

  if (do_rope) {
    const float* cr = cos_t + (size_t)s * D_;
    const float* sr = sin_t + (size_t)s * D_;
    float o[8];
#pragma unroll
    for (int j = 0; j < 8; ++j) {
      float p   = __shfl_xor(xn[j], 16, 32);       // element d ^ 128
      float rot = (lane < 16) ? -p : p;            // rot = [-x_hi, x_lo]
      o[j] = xn[j] * cr[d0 + j] + rot * sr[d0 + j];
    }
    *(float4*)(dst + d0)     = *(float4*)(o);
    *(float4*)(dst + d0 + 4) = *(float4*)(o + 4);
  } else {
    *(float4*)(dst + d0)     = *(float4*)(xn);
    *(float4*)(dst + d0 + 4) = *(float4*)(xn + 4);
  }
}

// ---------------------------------------------------------------------------
// Attention logits: logits[kv][n][l] = sum_d q_g[kv][n][d] * Krolled[kv][l][d]
// One wave per 16x32 tile (32 keys), fully unrolled D-loop -> 16 WMMAs/wave.
// Rolled cache: l < L-S -> cache_k[kv][l+S], else k_new[kv][l-(L-S)].
// B operand: each lane reads 16 contiguous floats of one key row (coalesced;
// db*4 <= 896B folds into the load immediate).
// ---------------------------------------------------------------------------
__global__ __launch_bounds__(32) void attn_logits_kernel(
    const float* __restrict__ q_g, const float* __restrict__ cache_k,
    const float* __restrict__ k_new, float* __restrict__ logits) {
  const int lane    = threadIdx.x & 31;
  const int keytile = blockIdx.x, rowtile = blockIdx.y, kv = blockIdx.z;
  const int base8   = (lane < 16) ? 0 : 16;
  const int koff    = (lane < 16) ? 0 : 16;
  const int keyn0   = keytile * 32 + (lane & 15);
  const int keyn1   = keyn0 + 16;

  const float* arow = q_g + (size_t)(kv * (G_ * S_) + rowtile * 16 + (lane & 15)) * D_ + base8;
  const float* krow0 = ((keyn0 < L_ - S_)
      ? cache_k + ((size_t)kv * L_ + keyn0 + S_) * D_
      : k_new   + ((size_t)kv * S_ + (keyn0 - (L_ - S_))) * D_) + koff;
  const float* krow1 = ((keyn1 < L_ - S_)
      ? cache_k + ((size_t)kv * L_ + keyn1 + S_) * D_
      : k_new   + ((size_t)kv * S_ + (keyn1 - (L_ - S_))) * D_) + koff;

  v8f acc0 = {0.f, 0.f, 0.f, 0.f, 0.f, 0.f, 0.f, 0.f};
  v8f acc1 = acc0;
#pragma unroll
  for (int db = 0; db < D_; db += 32) {
    float af[16], bf0[16], bf1[16];
    *(float4*)(af + 0)   = *(const float4*)(arow + db + 0);
    *(float4*)(af + 4)   = *(const float4*)(arow + db + 4);
    *(float4*)(af + 8)   = *(const float4*)(arow + db + 8);
    *(float4*)(af + 12)  = *(const float4*)(arow + db + 12);
#pragma unroll
    for (int j = 0; j < 16; j += 4) {
      *(float4*)(bf0 + j) = *(const float4*)(krow0 + db + j);
      *(float4*)(bf1 + j) = *(const float4*)(krow1 + db + j);
    }
    v16bf a = cvt16(af);
    acc0 = wmma_bf16(a, cvt16(bf0), acc0);
    acc1 = wmma_bf16(a, cvt16(bf1), acc1);
  }
  const int rbase = rowtile * 16 + ((lane >> 4) << 3);
  float* optr = logits + ((size_t)kv * (G_ * S_) + rbase) * L_ + keytile * 32 + (lane & 15);
#pragma unroll
  for (int r = 0; r < 8; ++r) {              // r*L folds into imm offset
    optr[r * L_]      = acc0[r];
    optr[r * L_ + 16] = acc1[r];
  }
}

// ---------------------------------------------------------------------------
// Row softmax over L=8192 (+mask), in place. One 256-thread block per row.
// ---------------------------------------------------------------------------
__global__ __launch_bounds__(256) void softmax_rows_kernel(
    float* __restrict__ logits, const float* __restrict__ mask) {
  __shared__ float red[256];
  const int r = blockIdx.x;            // 0..255  ( kv*64 + g*32 + s )
  const int t = threadIdx.x;
  const int s = r & 31;                // token index within new sequence
  float* row = logits + (size_t)r * L_;
  const float* mrow = mask + (size_t)s * L_;

  float x[32];
#pragma unroll
  for (int j = 0; j < 32; ++j) x[j] = row[t + j * 256] + mrow[t + j * 256];

  float mx = -3.4e38f;
#pragma unroll
  for (int j = 0; j < 32; ++j) mx = fmaxf(mx, x[j]);
  red[t] = mx; __syncthreads();
  for (int st = 128; st; st >>= 1) {
    if (t < st) red[t] = fmaxf(red[t], red[t + st]);
    __syncthreads();
  }
  mx = red[0]; __syncthreads();

  float sum = 0.f;
#pragma unroll
  for (int j = 0; j < 32; ++j) { x[j] = __expf(x[j] - mx); sum += x[j]; }
  red[t] = sum; __syncthreads();
  for (int st = 128; st; st >>= 1) {
    if (t < st) red[t] += red[t + st];
    __syncthreads();
  }
  const float inv = 1.0f / red[0];
#pragma unroll
  for (int j = 0; j < 32; ++j) row[t + j * 256] = x[j] * inv;
}

// ---------------------------------------------------------------------------
// AV: out[kv][n][d] = sum_l P[kv][n][l] * Vrolled[kv][l][d].
// One wave per 16x32 output tile (2 WMMAs/step sharing the P operand).
// V columns are strided in HBM, so stage a 32x32 f32 chunk through LDS
// (coalesced row loads), read columns, convert to bf16 B operands.
// Writes straight into attn_flat[S][H*D] layout for the output projection.
// ---------------------------------------------------------------------------
__global__ __launch_bounds__(32) void attn_av_kernel(
    const float* __restrict__ probs, const float* __restrict__ cache_v,
    const float* __restrict__ v_new, float* __restrict__ attn_flat) {
  __shared__ float vbuf[32 * 32];
  const int lane    = threadIdx.x & 31;
  const int dtile   = blockIdx.x, rowtile = blockIdx.y, kv = blockIdx.z;
  const int base8   = (lane < 16) ? 0 : 16;
  const int koff    = (lane < 16) ? 0 : 16;
  const float* arow = probs + (size_t)(kv * (G_ * S_) + rowtile * 16 + (lane & 15)) * L_ + base8;

  v8f acc0 = {0.f, 0.f, 0.f, 0.f, 0.f, 0.f, 0.f, 0.f};
  v8f acc1 = acc0;
  for (int lb = 0; lb < L_; lb += 32) {
    const int l = lb + lane;                        // this lane stages V row l
    const float* vrow = (l < L_ - S_)
        ? cache_v + ((size_t)kv * L_ + l + S_) * D_ + dtile * 32
        : v_new   + ((size_t)kv * S_ + (l - (L_ - S_))) * D_ + dtile * 32;
#pragma unroll
    for (int j = 0; j < 32; j += 4)
      *(float4*)(vbuf + lane * 32 + j) = *(const float4*)(vrow + j);
    __syncthreads();

    float af[16], bf0[16], bf1[16];
    *(float4*)(af + 0)  = *(const float4*)(arow + lb + 0);
    *(float4*)(af + 4)  = *(const float4*)(arow + lb + 4);
    *(float4*)(af + 8)  = *(const float4*)(arow + lb + 8);
    *(float4*)(af + 12) = *(const float4*)(arow + lb + 12);
#pragma unroll
    for (int j = 0; j < 16; ++j) {                  // imm-offset ds_load_b32
      bf0[j] = vbuf[(koff + j) * 32 + (lane & 15)];
      bf1[j] = vbuf[(koff + j) * 32 + 16 + (lane & 15)];
    }
    v16bf a = cvt16(af);
    acc0 = wmma_bf16(a, cvt16(bf0), acc0);
    acc1 = wmma_bf16(a, cvt16(bf1), acc1);
    __syncthreads();
  }

  const int rbase = rowtile * 16 + ((lane >> 4) << 3);
#pragma unroll
  for (int r = 0; r < 8; ++r) {
    const int nrow = rbase + r;                     // 0..63 within kv group
    const int s = nrow & 31, g = nrow >> 5;
    const int h = kv * G_ + g;
    float* orow = attn_flat + (size_t)s * (H_ * D_) + h * D_ + dtile * 32;
    orow[(lane & 15)]      = acc0[r];
    orow[(lane & 15) + 16] = acc1[r];
  }
}

// ---------------------------------------------------------------------------
// Host-side launch. Workspace (f32 elements):
//   q_lin 64K | k_lin 32K | v_lin 32K | q_g 64K | k_new 32K | v_new 32K
//   logits 2M (8MB, reused in-place as softmax probs) | attn_flat 64K
// total ~9.5 MB.
// ---------------------------------------------------------------------------
extern "C" void kernel_launch(void* const* d_in, const int* in_sizes, int n_in,
                              void* d_out, int out_size, void* d_ws, size_t ws_size,
                              hipStream_t stream) {
  const float* hidden  = (const float*)d_in[0];
  const float* cos_t   = (const float*)d_in[1];
  const float* sin_t   = (const float*)d_in[2];
  const float* cache_k = (const float*)d_in[3];
  const float* cache_v = (const float*)d_in[4];
  const float* mask    = (const float*)d_in[5];
  const float* W_q     = (const float*)d_in[6];
  const float* W_k     = (const float*)d_in[7];
  const float* W_v     = (const float*)d_in[8];
  const float* W_o     = (const float*)d_in[9];
  const float* qw      = (const float*)d_in[10];
  const float* kw      = (const float*)d_in[11];
  const float* vw      = (const float*)d_in[12];

  float* ws        = (float*)d_ws;
  float* q_lin     = ws;                               // [32][2048]
  float* k_lin     = q_lin + S_ * H_ * D_;             // [32][1024]
  float* v_lin     = k_lin + S_ * KV_ * D_;            // [32][1024]
  float* q_g       = v_lin + S_ * KV_ * D_;            // [4][64][256]
  float* k_new     = q_g   + KV_ * G_ * S_ * D_;       // [4][32][256]
  float* v_new     = k_new + KV_ * S_ * D_;            // [4][32][256]
  float* logits    = v_new + KV_ * S_ * D_;            // [4][64][8192]
  float* attn_flat = logits + (size_t)KV_ * G_ * S_ * L_;  // [32][2048]

  // QKV projections (bf16 WMMA GEMM, M=32, 16x32 tiles, compile-time shapes)
  gemm_bf16wmma<H_ * D_, HID_><<<dim3((H_ * D_) / 32, S_ / 16), 32, 0, stream>>>(hidden, W_q, q_lin);
  gemm_bf16wmma<KV_ * D_, HID_><<<dim3((KV_ * D_) / 32, S_ / 16), 32, 0, stream>>>(hidden, W_k, k_lin);
  gemm_bf16wmma<KV_ * D_, HID_><<<dim3((KV_ * D_) / 32, S_ / 16), 32, 0, stream>>>(hidden, W_v, v_lin);

  // RMSNorm + RoPE, regroup q to [KV][G*S][D], stage new k/v
  norm_rope_kernel<<<64, 256, 0, stream>>>(q_lin, k_lin, v_lin, cos_t, sin_t,
                                           qw, kw, vw, q_g, k_new, v_new);

  // q @ K_rolled^T  ->  logits [4][64][8192]
  attn_logits_kernel<<<dim3(L_ / 32, (G_ * S_) / 16, KV_), 32, 0, stream>>>(
      q_g, cache_k, k_new, logits);

  // softmax rows (in place)
  softmax_rows_kernel<<<KV_ * G_ * S_, 256, 0, stream>>>(logits, mask);

  // P @ V_rolled -> attn_flat [32][2048]
  attn_av_kernel<<<dim3(D_ / 32, (G_ * S_) / 16, KV_), 32, 0, stream>>>(
      logits, cache_v, v_new, attn_flat);

  // output projection -> d_out [32][2560]
  gemm_bf16wmma<HID_, H_ * D_><<<dim3(HID_ / 32, S_ / 16), 32, 0, stream>>>(
      attn_flat, W_o, (float*)d_out);
}
